// Head_11544872091814
// MI455X (gfx1250) — matrix operand order repaired and checked by reading, not compiled, not run on hardware
//
#include <hip/hip_runtime.h>
#include <hip/hip_bf16.h>

// Flash attention for CDNA5 / gfx1250 (wave32, WMMA 16x16x32 f16 -> f32).
//
// Pipeline (all on `stream`):
//   1) cvt_kernel : x, Wq/Wk/Wv  fp32 -> f16  (into d_ws)
//   2) proj_kernel: q = x Wq^T, k = x Wk^T, vT = (x Wv^T)^T   (WMMA, f16 out)
//   3) attn_kernel: causal flash attention, one wave per 16-row query tile
//
// Workspace (f16 elements):
//   x16[ B*T*C ] | w16[ 3*H*C ] | q16[ B*T*H ] | k16[ B*T*H ] | vT16[ B*H*T ]
//   total = 16,826,368 f16 = ~33.7 MB  (assumed <= ws_size)

#define B_ 16
#define T_ 2048
#define C_ 128
#define H_ 128

typedef __attribute__((ext_vector_type(8)))  _Float16 v8h;
typedef __attribute__((ext_vector_type(16))) _Float16 v16h;
typedef __attribute__((ext_vector_type(8)))  float    v8f;

static constexpr size_t NX    = (size_t)B_ * T_ * C_;   // 4,194,304
static constexpr size_t NW    = (size_t)3 * H_ * C_;    // 49,152
static constexpr size_t OFF_X = 0;
static constexpr size_t OFF_W = NX;
static constexpr size_t OFF_Q = OFF_W + NW;
static constexpr size_t OFF_K = OFF_Q + (size_t)B_ * T_ * H_;
static constexpr size_t OFF_V = OFF_K + (size_t)B_ * T_ * H_;

__device__ __forceinline__ v16h cat8(v8h lo, v8h hi) {
  return __builtin_shufflevector(lo, hi, 0,1,2,3,4,5,6,7,8,9,10,11,12,13,14,15);
}
// B-fragment (32x16, 16-bit): lane holds two 8-f16 chunks 16 apart in K
// (lanes 0-15: K {0..7,16..23}; lanes 16-31: K {8..15,24..31}; caller applies +8).
__device__ __forceinline__ v16h load_bfrag(const _Float16* p) {
  return cat8(*(const v8h*)p, *(const v8h*)(p + 16));
}
// A-fragment (16x32, 16-bit): lane holds two 8-f16 chunks 16 apart in K
// (lanes 0-15: K {0..7,16..23}; lanes 16-31: K {8..15,24..31}; caller applies +8).
__device__ __forceinline__ v16h load_afrag(const _Float16* p) {
  return cat8(*(const v8h*)p, *(const v8h*)(p + 16));
}
__device__ __forceinline__ v8f wmma_f16(v16h a, v16h b, v8f c) {
  return __builtin_amdgcn_wmma_f32_16x16x32_f16(false, a, false, b, (short)0, c,
                                                false, false);
}

// ---------------------------------------------------------------- convert
__global__ void cvt_kernel(const float* __restrict__ x,
                           const float* __restrict__ Wq,
                           const float* __restrict__ Wk,
                           const float* __restrict__ Wv,
                           _Float16* __restrict__ h) {
  size_t i = (size_t)blockIdx.x * blockDim.x + threadIdx.x;
  if (i < NX) {
    h[OFF_X + i] = (_Float16)x[i];
  } else if (i < NX + NW) {
    size_t j = i - NX;
    int mat = (int)(j >> 14);          // 16384 = H_*C_
    int o   = (int)(j & 16383);
    const float* s = (mat == 0) ? Wq : ((mat == 1) ? Wk : Wv);
    h[OFF_W + j] = (_Float16)s[o];
  }
}

// ---------------------------------------------------------------- projection
// One wave computes one 16x128 output tile of one of {q, k, v}.
__global__ __launch_bounds__(128) void proj_kernel(
    const _Float16* __restrict__ x16, const _Float16* __restrict__ w16,
    _Float16* __restrict__ q16, _Float16* __restrict__ k16,
    _Float16* __restrict__ vT16) {
  const int lane  = threadIdx.x & 31;
  const int wave  = threadIdx.x >> 5;
  const int ntile = B_ * T_ / 16;                 // 2048 row tiles
  const int task  = blockIdx.x * 4 + wave;        // 0..6143
  const int mat   = task / ntile;                 // 0=q 1=k 2=v
  const int tile  = task % ntile;
  const int laneN = lane & 15;
  const int off8  = (lane >= 16) ? 8 : 0;
  const int off16 = (lane >= 16) ? 16 : 0;

  // A-fragments of x tile (16 rows x 128, 4 K-steps of 32)
  v16h a[4];
  {
    const _Float16* row = x16 + (size_t)(tile * 16 + laneN) * C_;
#pragma unroll
    for (int kk = 0; kk < 4; ++kk)
      a[kk] = load_afrag(row + kk * 32 + off8);
  }

  const _Float16* Wm = w16 + (size_t)mat * H_ * C_;
#pragma unroll
  for (int nt = 0; nt < 8; ++nt) {
    v8f acc = {};
#pragma unroll
    for (int kk = 0; kk < 4; ++kk) {
      // B[c,h] = W[h,c]: column h of B is W row h, contiguous along c.
      const _Float16* p = Wm + (size_t)(nt * 16 + laneN) * C_ + kk * 32 + off8;
      acc = wmma_f16(a[kk], load_bfrag(p), acc);
    }
    // C/D layout: VGPR r -> row M = r + off8, col N = laneN
    if (mat == 2) {
      const int bb = tile / (T_ / 16);
      const int t0 = (tile % (T_ / 16)) * 16;
      const int hc = nt * 16 + laneN;
#pragma unroll
      for (int r = 0; r < 8; ++r)
        vT16[((size_t)bb * H_ + hc) * T_ + (t0 + r + off8)] = (_Float16)acc[r];
    } else {
      _Float16* dst = (mat == 0) ? q16 : k16;
#pragma unroll
      for (int r = 0; r < 8; ++r)
        dst[(size_t)(tile * 16 + r + off8) * H_ + nt * 16 + laneN] =
            (_Float16)acc[r];
    }
  }
}

// ---------------------------------------------------------------- attention
// One wave per 16-row query tile. Key blocks of 32 -> two 16x16 score tiles,
// online softmax, P (16x32) staged through wave-private LDS, O += P*V.
__global__ __launch_bounds__(128) void attn_kernel(
    const _Float16* __restrict__ q16, const _Float16* __restrict__ k16,
    const _Float16* __restrict__ vT16, float* __restrict__ out) {
  __shared__ _Float16 p_lds[4][16][32];           // per-wave 16x32 P tile

  const int lane  = threadIdx.x & 31;
  const int wave  = threadIdx.x >> 5;
  const int tile  = blockIdx.x * 4 + wave;        // 0..2047
  const int bb    = tile / (T_ / 16);
  const int qt    = tile % (T_ / 16);
  const int laneN = lane & 15;
  const int off8  = (lane >= 16) ? 8 : 0;
  const int off16 = (lane >= 16) ? 16 : 0;
  const float sc  = 0.08838834764831845f;         // C^-0.5

  const _Float16* qb = q16 + (size_t)bb * T_ * H_;
  const _Float16* kb = k16 + (size_t)bb * T_ * H_;
  const _Float16* vb = vT16 + (size_t)bb * H_ * T_;

  // Q A-fragments (resident whole kernel)
  v16h qa[4];
  {
    const _Float16* row = qb + (size_t)(qt * 16 + laneN) * H_;
#pragma unroll
    for (int kk = 0; kk < 4; ++kk)
      qa[kk] = load_afrag(row + kk * 32 + off8);
  }

  v8f  acc[8];
  float m[8], l[8];
  {
    v8f z = {};
#pragma unroll
    for (int n = 0; n < 8; ++n) acc[n] = z;
#pragma unroll
    for (int r = 0; r < 8; ++r) { m[r] = -__builtin_inff(); l[r] = 0.0f; }
  }

  const int nkb = qt / 2 + 1;                     // causal: key blocks of 32
  for (int kbI = 0; kbI < nkb; ++kbI) {
    const int ks = kbI * 32;

    // S = Q K^T : two 16x16 tiles (keys ks..+15, ks+16..+31)
    v8f s0 = {}, s1 = {};
    const _Float16* kr0 = kb + (size_t)(ks + laneN) * H_;
    const _Float16* kr1 = kb + (size_t)(ks + 16 + laneN) * H_;
#pragma unroll
    for (int kk = 0; kk < 4; ++kk) {
      s0 = wmma_f16(qa[kk], load_bfrag(kr0 + kk * 32 + off8), s0);
      s1 = wmma_f16(qa[kk], load_bfrag(kr1 + kk * 32 + off8), s1);
    }

    // scale + causal mask + online softmax; stage P(f16) into LDS
#pragma unroll
    for (int r = 0; r < 8; ++r) {
      const int qrow = qt * 16 + r + off8;
      float x0 = s0[r] * sc; if (ks + laneN > qrow)      x0 = -__builtin_inff();
      float x1 = s1[r] * sc; if (ks + 16 + laneN > qrow) x1 = -__builtin_inff();
      float mx = fmaxf(x0, x1);
#pragma unroll
      for (int d = 1; d < 16; d <<= 1)
        mx = fmaxf(mx, __shfl_xor(mx, d, 32));    // xor<16 stays in half-wave
      const float mn  = fmaxf(m[r], mx);
      const float rsc = __expf(m[r] - mn);
      m[r] = mn;
      const float p0 = __expf(x0 - mn);
      const float p1 = __expf(x1 - mn);
      float rs = p0 + p1;
#pragma unroll
      for (int d = 1; d < 16; d <<= 1)
        rs += __shfl_xor(rs, d, 32);
      l[r] = l[r] * rsc + rs;
      p_lds[wave][r + off8][laneN]      = (_Float16)p0;
      p_lds[wave][r + off8][laneN + 16] = (_Float16)p1;
#pragma unroll
      for (int n = 0; n < 8; ++n) acc[n][r] *= rsc;
    }

    // C-layout -> A-layout reshape via wave-private LDS (DS in-order per wave)
    const v16h pa = load_afrag(&p_lds[wave][laneN][off8]);

    // O += P * V  (B[s,h] = vT[h][s], contiguous along s)
#pragma unroll
    for (int n = 0; n < 8; ++n) {
      const _Float16* vp = vb + (size_t)(n * 16 + laneN) * T_ + ks + off8;
      acc[n] = wmma_f16(pa, load_bfrag(vp), acc[n]);
    }
  }

  // epilogue: out = acc / l   (fp32)
  float* ob = out + ((size_t)bb * T_ + qt * 16) * H_;
#pragma unroll
  for (int r = 0; r < 8; ++r) {
    const float inv = 1.0f / l[r];
#pragma unroll
    for (int n = 0; n < 8; ++n)
      ob[(size_t)(r + off8) * H_ + n * 16 + laneN] = acc[n][r] * inv;
  }
}

// ---------------------------------------------------------------- launch
extern "C" void kernel_launch(void* const* d_in, const int* in_sizes, int n_in,
                              void* d_out, int out_size, void* d_ws,
                              size_t ws_size, hipStream_t stream) {
  (void)in_sizes; (void)n_in; (void)out_size; (void)ws_size;
  const float* x  = (const float*)d_in[0];
  const float* Wq = (const float*)d_in[1];
  const float* Wk = (const float*)d_in[2];
  const float* Wv = (const float*)d_in[3];
  float* out = (float*)d_out;

  _Float16* h   = (_Float16*)d_ws;
  _Float16* x16 = h + OFF_X;
  _Float16* w16 = h + OFF_W;
  _Float16* q16 = h + OFF_Q;
  _Float16* k16 = h + OFF_K;
  _Float16* v16 = h + OFF_V;

  const size_t totalCvt = NX + NW;
  cvt_kernel<<<(unsigned)((totalCvt + 255) / 256), 256, 0, stream>>>(
      x, Wq, Wk, Wv, h);
  proj_kernel<<<(3 * (B_ * T_ / 16)) / 4, 128, 0, stream>>>(
      x16, w16, q16, k16, v16);
  attn_kernel<<<(B_ * T_ / 16) / 4, 128, 0, stream>>>(q16, k16, v16, out);
}
